// QuantumFractalAttention_8529805050313
// MI455X (gfx1250) — matrix operand order repaired and checked by reading, not compiled, not run on hardware
//
#include <hip/hip_runtime.h>

// ---------------- problem constants ----------------
#define B_   2
#define T_   2048
#define D_   1024
#define H_   16
#define DK_  64
#define MT_  (B_ * T_)          // 4096 rows

#define USE_TDM 1

typedef __attribute__((ext_vector_type(16))) __bf16 v16bf;
typedef __attribute__((ext_vector_type(8)))  float  v8f;
typedef __attribute__((ext_vector_type(4)))  unsigned int v4u;
typedef __attribute__((ext_vector_type(8)))  int    v8i;
typedef __attribute__((ext_vector_type(4)))  int    v4i;

// ---------------- wave32 WMMA fragment helpers ----------------
// Layouts used by these helpers (wave32, contiguous k for A and B):
//  A (16x32 bf16): lane L: M=L&15, half=L>>4; elems 0..7 -> K=half*16+j,
//                  elems 8..15 -> K=half*16+8+j
//  B (32x16 bf16): lane L: N=L&15, half=L>>4; elem e -> K=half*16+e
//  C/D (16x16 f32): lane L: N=L&15, half=L>>4; elem v -> M=v+8*half

__device__ inline v8f v8f_splat(float s) {
    v8f r;
#pragma unroll
    for (int i = 0; i < 8; ++i) r[i] = s;
    return r;
}

__device__ inline v16bf load_a_bf16(const __bf16* __restrict__ src, int ld,
                                    int m0, int k0, int lane) {
    const int half = lane >> 4;
    const __bf16* row = src + (size_t)(m0 + (lane & 15)) * ld + k0;
    v16bf a;
#pragma unroll
    for (int j = 0; j < 8; ++j) a[j]     = row[half * 16 + j];
#pragma unroll
    for (int j = 0; j < 8; ++j) a[8 + j] = row[half * 16 + 8 + j];
    return a;
}

// B[k][n] = W[n][k] (for y = x @ W^T): lane n reads 32B contiguous of W row n
__device__ inline v16bf load_b_bf16(const __bf16* __restrict__ W, int n,
                                    int k0, int half) {
    const __bf16* row = W + (size_t)n * D_ + k0 + half * 16;
    v16bf b;
#pragma unroll
    for (int e = 0; e < 16; ++e) b[e] = row[e];
    return b;
}

// ---------------- pass 0: fp32 -> bf16 conversion ----------------
__global__ __launch_bounds__(256) void cvt_kernel(const float* __restrict__ src,
                                                  __bf16* __restrict__ dst, int n) {
    const int i = (blockIdx.x * 256 + threadIdx.x) * 8;
    if (i + 7 < n) {
#pragma unroll
        for (int j = 0; j < 8; ++j) dst[i + j] = (__bf16)src[i + j];
    }
}

// ---------------- shared GEMM core: 32x64 tile per wave, K=1024 ----------------
// acc[r][c]: r = row subtile (16 rows each), c = 16-col subtile.
// Double-buffered fragment loads to hide global latency behind WMMA.
__device__ inline void gemm_core_32x64(const __bf16* __restrict__ A,
                                       const __bf16* __restrict__ W,
                                       int m0, int n0, int lane, v8f acc[2][4]) {
    const int half = lane >> 4, lc = lane & 15;
    v16bf a0 = load_a_bf16(A, D_, m0, 0, lane);
    v16bf a1 = load_a_bf16(A, D_, m0 + 16, 0, lane);
    v16bf bfr[4];
#pragma unroll
    for (int c = 0; c < 4; ++c) bfr[c] = load_b_bf16(W, n0 + c * 16 + lc, 0, half);

    for (int kk = 0; kk < D_; kk += 32) {
        const int kn = (kk + 32 < D_) ? (kk + 32) : 0;   // last-iter dummy reload
        v16bf na0 = load_a_bf16(A, D_, m0, kn, lane);
        v16bf na1 = load_a_bf16(A, D_, m0 + 16, kn, lane);
        v16bf nb[4];
#pragma unroll
        for (int c = 0; c < 4; ++c) nb[c] = load_b_bf16(W, n0 + c * 16 + lc, kn, half);

#pragma unroll
        for (int c = 0; c < 4; ++c) {
            acc[0][c] = __builtin_amdgcn_wmma_f32_16x16x32_bf16(
                false, a0, false, bfr[c], (short)0, acc[0][c], false, false);
            acc[1][c] = __builtin_amdgcn_wmma_f32_16x16x32_bf16(
                false, a1, false, bfr[c], (short)0, acc[1][c], false, false);
        }
        a0 = na0; a1 = na1;
#pragma unroll
        for (int c = 0; c < 4; ++c) bfr[c] = nb[c];
    }
}

// ---------------- kernel 1: fused QKV projection ----------------
// grid = (256, 1, 3), block = 256 (8 waves). Wave tile: 32x64 of y = x@W^T+b.
// Epilogue: noise (Q,K), 1/sqrt(DK) folded into Q, bf16 store to [B,H,T,DK].
__global__ __launch_bounds__(256) void qkv_proj_kernel(
    const __bf16* __restrict__ xb,
    const __bf16* __restrict__ wqb, const float* __restrict__ bq,
    const __bf16* __restrict__ wkb, const float* __restrict__ bk,
    const __bf16* __restrict__ wvb, const float* __restrict__ bv,
    const float* __restrict__ qr, const float* __restrict__ qi,
    const float* __restrict__ kr, const float* __restrict__ ki,
    __bf16* __restrict__ qb, __bf16* __restrict__ kb, __bf16* __restrict__ vb)
{
    const int sel = blockIdx.z;                     // 0=Q 1=K 2=V
    const __bf16* W   = (sel == 0) ? wqb : (sel == 1) ? wkb : wvb;
    const float* bias = (sel == 0) ? bq  : (sel == 1) ? bk  : bv;
    const float* nr   = (sel == 0) ? qr  : (sel == 1) ? kr  : nullptr;
    const float* ni   = (sel == 0) ? qi  : (sel == 1) ? ki  : nullptr;
    __bf16* out       = (sel == 0) ? qb  : (sel == 1) ? kb  : vb;
    const float oscale = (sel == 0) ? 0.125f : 1.0f;

    const int lane = threadIdx.x & 31;
    const int wave = threadIdx.x >> 5;
    const int wt   = blockIdx.x * 8 + wave;          // 0..2047
    const int m0   = (wt >> 4) * 32;
    const int n0   = (wt & 15) * 64;
    const int half = lane >> 4, lc = lane & 15;

    v8f acc[2][4];
#pragma unroll
    for (int r = 0; r < 2; ++r)
#pragma unroll
        for (int c = 0; c < 4; ++c) acc[r][c] = v8f_splat(bias[n0 + c * 16 + lc]);

    gemm_core_32x64(xb, W, m0, n0, lane, acc);

    const int b = m0 / T_;                           // tile never straddles batch
#pragma unroll
    for (int r = 0; r < 2; ++r) {
#pragma unroll
        for (int c = 0; c < 4; ++c) {
            const int n  = n0 + c * 16 + lc;
            const int hh = n >> 6, dk = n & 63;
#pragma unroll
            for (int v = 0; v < 8; ++v) {
                const int m = m0 + r * 16 + v + 8 * half;
                const int t = m - b * T_;
                float val = acc[r][c][v];
                const size_t idx = (((size_t)(b * H_ + hh)) * T_ + t) * DK_ + dk;
                if (nr) {
                    float re = nr[idx];
                    float im = ni[idx];
                    val += im * rsqrtf(re * re + im * im + 1e-30f); // sin(atan2)
                }
                out[idx] = (__bf16)(val * oscale);
            }
        }
    }
}

// ---------------- TDM: stage K tile (32 x 64 bf16) into padded LDS ----------------
#if USE_TDM
// D# per CDNA5 ISA. Row-major 2D tile, data_size=2B,
// pad_enable: +1 DWORD after every 32 DWORDs (128B row) -> LDS row pitch 66 bf16.
__device__ inline void tdm_load_tile_bf16(const __bf16* gsrc, unsigned lds_off,
                                          int rows) {
    const unsigned long long ga = (unsigned long long)(uintptr_t)gsrc;
    v4u g0;
    g0[0] = 1u;                                        // count=1, user mode
    g0[1] = lds_off;                                   // lds_addr
    g0[2] = (unsigned)(ga & 0xffffffffu);              // global_addr[31:0]
    g0[3] = (unsigned)((ga >> 32) & 0x1ffffffu)        // global_addr[56:32]
          | (2u << 30);                                // type=2 (image)
    v8i g1;
    g1[0] = (1 << 16)                                  // data_size=1 -> 2 bytes
          | (1 << 20)                                  // pad_enable
          | (4 << 22)                                  // pad_interval: 32 DWORDs
          | (0 << 25);                                 // pad_amount: 1 DWORD
    g1[1] = (DK_ & 0xffff) << 16;                      // tensor_dim0 = 64
    g1[2] = (T_ & 0xffff) << 16;                       // tensor_dim1 = 2048
    g1[3] = (DK_ & 0xffff) << 16;                      // tile_dim0 = 64
    g1[4] = rows & 0xffff;                             // tile_dim1 = 32
    g1[5] = DK_;                                       // tensor_dim0_stride = 64
    g1[6] = 0;
    g1[7] = 0;
    v4i z4 = {0, 0, 0, 0};
#if __has_include(<hip/amd_detail/amd_gfx1250_TDM.h>)
    v8i z8 = {0, 0, 0, 0, 0, 0, 0, 0};
    __builtin_amdgcn_tensor_load_to_lds(g0, g1, z4, z4, z8, 0);
#else
    __builtin_amdgcn_tensor_load_to_lds(g0, g1, z4, z4, 0);
#endif
}
#endif

// ---------------- kernel 2: flash attention per (b,h) ----------------
// grid = (T/64, B*H), block = 128 (4 waves). K tile staged by TDM (shared by
// all 4 waves), V transposed into LDS, online softmax, P@V via WMMA.
#define KT_PITCH 66   // bf16 elements per padded LDS row (33 DWORDs)

__global__ __launch_bounds__(128) void attn_kernel(
    const __bf16* __restrict__ qb, const __bf16* __restrict__ kb,
    const __bf16* __restrict__ vb, __bf16* __restrict__ ab)
{
    __shared__ __bf16 KT[32 * KT_PITCH]; // K tile, row-major, padded pitch
    __shared__ __bf16 VT[64][34];        // V^T tile: [d][s_local], padded
    __shared__ __bf16 Pw[4][16][32];     // per-wave P tile (C->A relayout)

    const int tid  = threadIdx.x;
    const int lane = tid & 31;
    const int wave = tid >> 5;
    const int half = lane >> 4, lc = lane & 15;
    const int bh = blockIdx.y;
    const int b  = bh >> 4, h = bh & 15;
    const int t0 = blockIdx.x * 64 + wave * 16;

    const __bf16* Q = qb + (size_t)bh * T_ * DK_;
    const __bf16* K = kb + (size_t)bh * T_ * DK_;
    const __bf16* V = vb + (size_t)bh * T_ * DK_;

#if USE_TDM
    const unsigned kt_off =
        (unsigned)(unsigned long long)(__attribute__((address_space(3))) void*)&KT[0];
#endif

    v16bf qf[2];
#pragma unroll
    for (int kq = 0; kq < 2; ++kq)
        qf[kq] = load_a_bf16(Q, DK_, t0, kq * 32, lane);

    v8f acc[4];
#pragma unroll
    for (int dt = 0; dt < 4; ++dt) acc[dt] = v8f_splat(0.0f);
    float ms[8], ls[8];
#pragma unroll
    for (int v = 0; v < 8; ++v) { ms[v] = -1e30f; ls[v] = 0.0f; }

    for (int s0 = 0; s0 < T_; s0 += 32) {
        __syncthreads();                         // protect previous tile reads
#if USE_TDM
        if (wave == 0)                           // one async TDM issue per block
            tdm_load_tile_bf16(K + (size_t)s0 * DK_, kt_off, 32);
#else
        {   // cooperative K stage with padded pitch
            const int sl = tid & 31, dg = tid >> 5;
            const __bf16* krow = K + (size_t)(s0 + sl) * DK_ + dg * 16;
#pragma unroll
            for (int i = 0; i < 16; ++i) KT[sl * KT_PITCH + dg * 16 + i] = krow[i];
        }
#endif
        {   // cooperative transpose-stage of V[s0:s0+32][0:64] -> VT[d][s]
            const int sl = tid & 31, dg = tid >> 5;
            const __bf16* vrow = V + (size_t)(s0 + sl) * DK_ + dg * 16;
#pragma unroll
            for (int i = 0; i < 16; ++i) VT[dg * 16 + i][sl] = vrow[i];
        }
#if USE_TDM
        if (wave == 0) __builtin_amdgcn_s_wait_tensorcnt(0);
#endif
        __syncthreads();

        // S = (Q/sqrt(DK)) @ K^T from the LDS K tile
        v8f S[2];
#pragma unroll
        for (int ct = 0; ct < 2; ++ct) {
            v8f s_ = v8f_splat(0.0f);
#pragma unroll
            for (int kd = 0; kd < 2; ++kd) {
                const __bf16* krow =
                    &KT[(ct * 16 + lc) * KT_PITCH + kd * 32 + half * 16];
                v16bf kf;
#pragma unroll
                for (int e = 0; e < 16; ++e) kf[e] = krow[e];
                s_ = __builtin_amdgcn_wmma_f32_16x16x32_bf16(
                    false, qf[kd], false, kf, (short)0, s_, false, false);
            }
            S[ct] = s_;
        }

        // online softmax: rows live across the 16 lanes of each half
#pragma unroll
        for (int v = 0; v < 8; ++v) {
            float s0v = S[0][v], s1v = S[1][v];
            float mx = fmaxf(s0v, s1v);
#pragma unroll
            for (int msk = 1; msk < 16; msk <<= 1)
                mx = fmaxf(mx, __shfl_xor(mx, msk, 32));
            float mnew  = fmaxf(ms[v], mx);
            float scale = __expf(ms[v] - mnew);
            float p0 = __expf(s0v - mnew);
            float p1 = __expf(s1v - mnew);
            float rs = p0 + p1;
#pragma unroll
            for (int msk = 1; msk < 16; msk <<= 1)
                rs += __shfl_xor(rs, msk, 32);
            ls[v] = ls[v] * scale + rs;
            ms[v] = mnew;
#pragma unroll
            for (int dt = 0; dt < 4; ++dt) acc[dt][v] *= scale;
            Pw[wave][8 * half + v][lc]      = (__bf16)p0;
            Pw[wave][8 * half + v][16 + lc] = (__bf16)p1;
        }
        asm volatile("s_wait_dscnt 0" ::: "memory"); // intra-wave LDS RAW fence

        v16bf pf;                                    // P as A-fragment
        {
            const __bf16* prow = &Pw[wave][lc][0];
#pragma unroll
            for (int j = 0; j < 8; ++j) pf[j]     = prow[half * 16 + j];
#pragma unroll
            for (int j = 0; j < 8; ++j) pf[8 + j] = prow[half * 16 + 8 + j];
        }

#pragma unroll
        for (int dt = 0; dt < 4; ++dt) {             // acc += P @ V
            const __bf16* vtrow = &VT[dt * 16 + lc][half * 16];
            v16bf vf;
#pragma unroll
            for (int e = 0; e < 16; ++e) vf[e] = vtrow[e];
            acc[dt] = __builtin_amdgcn_wmma_f32_16x16x32_bf16(
                false, pf, false, vf, (short)0, acc[dt], false, false);
        }
    }

#pragma unroll
    for (int dt = 0; dt < 4; ++dt) {                 // normalize, store [B,T,D]
#pragma unroll
        for (int v = 0; v < 8; ++v) {
            const int t  = t0 + v + 8 * half;
            const int dc = dt * 16 + lc;
            float val = acc[dt][v] / ls[v];
            ab[((size_t)(b * T_ + t)) * D_ + h * 64 + dc] = (__bf16)val;
        }
    }
}

// ---------------- kernel 3: output projection ----------------
__global__ __launch_bounds__(256) void out_proj_kernel(
    const __bf16* __restrict__ ab, const __bf16* __restrict__ wob,
    const float* __restrict__ bo, float* __restrict__ out)
{
    const int lane = threadIdx.x & 31;
    const int wave = threadIdx.x >> 5;
    const int wt   = blockIdx.x * 8 + wave;
    const int m0   = (wt >> 4) * 32;
    const int n0   = (wt & 15) * 64;
    const int half = lane >> 4, lc = lane & 15;

    v8f acc[2][4];
#pragma unroll
    for (int r = 0; r < 2; ++r)
#pragma unroll
        for (int c = 0; c < 4; ++c) acc[r][c] = v8f_splat(bo[n0 + c * 16 + lc]);

    gemm_core_32x64(ab, wob, m0, n0, lane, acc);

#pragma unroll
    for (int r = 0; r < 2; ++r) {
#pragma unroll
        for (int c = 0; c < 4; ++c) {
            const int n = n0 + c * 16 + lc;
#pragma unroll
            for (int v = 0; v < 8; ++v) {
                const int m = m0 + r * 16 + v + 8 * half;
                out[(size_t)m * D_ + n] = acc[r][c][v];
            }
        }
    }
}

// ---------------- launcher ----------------
extern "C" void kernel_launch(void* const* d_in, const int* in_sizes, int n_in,
                              void* d_out, int out_size, void* d_ws, size_t ws_size,
                              hipStream_t stream) {
    const float* x  = (const float*)d_in[0];
    const float* Wq = (const float*)d_in[1];
    const float* bq = (const float*)d_in[2];
    const float* Wk = (const float*)d_in[3];
    const float* bk = (const float*)d_in[4];
    const float* Wv = (const float*)d_in[5];
    const float* bv = (const float*)d_in[6];
    const float* Wo = (const float*)d_in[7];
    const float* bo = (const float*)d_in[8];
    const float* qr = (const float*)d_in[9];
    const float* qi = (const float*)d_in[10];
    const float* kr = (const float*)d_in[11];
    const float* ki = (const float*)d_in[12];

    // workspace (bf16): x(4M) | Wq,Wk,Wv,Wo(1M each) | Q,K,V,attn(4M each) = 48MB
    __bf16* wsb = (__bf16*)d_ws;
    const size_t nX = (size_t)MT_ * D_;          // 4,194,304
    const size_t nW = (size_t)D_ * D_;           // 1,048,576
    __bf16* xb  = wsb;
    __bf16* wqb = xb  + nX;
    __bf16* wkb = wqb + nW;
    __bf16* wvb = wkb + nW;
    __bf16* wob = wvb + nW;
    __bf16* qbf = wob + nW;
    __bf16* kbf = qbf + nX;
    __bf16* vbf = kbf + nX;
    __bf16* abf = vbf + nX;

    // pass 0: one-time fp32 -> bf16 conversion (weights stay L2-resident)
    cvt_kernel<<<(int)(nX / 2048), 256, 0, stream>>>(x,  xb,  (int)nX);
    cvt_kernel<<<(int)(nW / 2048), 256, 0, stream>>>(Wq, wqb, (int)nW);
    cvt_kernel<<<(int)(nW / 2048), 256, 0, stream>>>(Wk, wkb, (int)nW);
    cvt_kernel<<<(int)(nW / 2048), 256, 0, stream>>>(Wv, wvb, (int)nW);
    cvt_kernel<<<(int)(nW / 2048), 256, 0, stream>>>(Wo, wob, (int)nW);

    qkv_proj_kernel<<<dim3(256, 1, 3), 256, 0, stream>>>(
        xb, wqb, bq, wkb, bk, wvb, bv, qr, qi, kr, ki, qbf, kbf, vbf);

    attn_kernel<<<dim3(T_ / 64, B_ * H_), 128, 0, stream>>>(qbf, kbf, vbf, abf);

    out_proj_kernel<<<dim3(256), 256, 0, stream>>>(abf, wob, bo, (float*)d_out);
}
